// ABNet_67250597920810
// MI455X (gfx1250) — matrix operand order repaired and checked by reading, not compiled, not run on hardware
//
#include <hip/hip_runtime.h>
#include <hip/hip_bf16.h>
#include <math.h>

#define HNUM  10
#define BTOT  32768
#define DDIM  256
#define MT    64            // batch rows per workgroup
#define PITCH 264           // f16 elements per LDS row (256 + 8 pad, keeps 16B alignment)

typedef __attribute__((ext_vector_type(16))) _Float16 v16h;
typedef __attribute__((ext_vector_type(8)))  _Float16 v8h;
typedef __attribute__((ext_vector_type(8)))  float    v8f;

static __device__ __forceinline__ v16h cat8(v8h lo, v8h hi) {
  return __builtin_shufflevector(lo, hi, 0,1,2,3,4,5,6,7,8,9,10,11,12,13,14,15);
}

// dst[64][256] = relu( src[64][256] @ W^T + b ), W row-major [out=256][in=256] f16.
// Wave w owns N-tiles {2w, 2w+1}; loops all 4 M-subtiles so each weight fragment
// is loaded once per workgroup-layer.
// Fragment layouts per CDNA5 ISA 7.12.2:
//   A 16x32 f16 : lane(m=L&15, hf=L>>4): VGPR0-3 K=hf*8+[0..7], VGPR4-7 K=16+hf*8+[0..7]
//   B 32x16 f16 : lane(n=L&15, hf=L>>4): VGPR0-3 K=hf*8+[0..7], VGPR4-7 K=16+hf*8+[0..7]  (rows of W)
//   C 16x16 f32 : lane L holds N=L&15, VGPR v holds M = v + 8*(L>>4)
static __device__ __forceinline__ void gemm_layer(
    const _Float16* __restrict__ src,      // LDS [MT][PITCH]
    const _Float16* __restrict__ W,        // global f16 [256][256]
    const float*    __restrict__ bias,     // global f32 [256]
    _Float16* __restrict__ dst)            // LDS [MT][PITCH]
{
  const int lane = threadIdx.x & 31;
  const int wv   = threadIdx.x >> 5;       // 0..7
  const int ml   = lane & 15;
  const int hf   = lane >> 4;
  #pragma unroll
  for (int nt2 = 0; nt2 < 2; ++nt2) {
    const int ntile = wv * 2 + nt2;        // 0..15
    const int n     = ntile * 16 + ml;
    const _Float16* wrow = W + (size_t)n * DDIM;
    v8f acc0 = {0.f,0.f,0.f,0.f,0.f,0.f,0.f,0.f};
    v8f acc1 = acc0, acc2 = acc0, acc3 = acc0;
    #pragma unroll
    for (int kb = 0; kb < 8; ++kb) {
      const _Float16* wp = wrow + kb * 32 + hf * 8;
      const v16h bfrag = cat8(*(const v8h*)wp, *(const v8h*)(wp + 16));
      const _Float16* ap = src + ml * PITCH + kb * 32 + hf * 8;
      const v16h a0 = cat8(*(const v8h*)(ap            ), *(const v8h*)(ap             + 16));
      const v16h a1 = cat8(*(const v8h*)(ap + 16*PITCH ), *(const v8h*)(ap + 16*PITCH  + 16));
      const v16h a2 = cat8(*(const v8h*)(ap + 32*PITCH ), *(const v8h*)(ap + 32*PITCH  + 16));
      const v16h a3 = cat8(*(const v8h*)(ap + 48*PITCH ), *(const v8h*)(ap + 48*PITCH  + 16));
      acc0 = __builtin_amdgcn_wmma_f32_16x16x32_f16(false, a0, false, bfrag, (short)0, acc0, false, false);
      acc1 = __builtin_amdgcn_wmma_f32_16x16x32_f16(false, a1, false, bfrag, (short)0, acc1, false, false);
      acc2 = __builtin_amdgcn_wmma_f32_16x16x32_f16(false, a2, false, bfrag, (short)0, acc2, false, false);
      acc3 = __builtin_amdgcn_wmma_f32_16x16x32_f16(false, a3, false, bfrag, (short)0, acc3, false, false);
    }
    const float bv = bias[n];
    v8f accs[4] = {acc0, acc1, acc2, acc3};
    #pragma unroll
    for (int ms = 0; ms < 4; ++ms) {
      #pragma unroll
      for (int v = 0; v < 8; ++v) {
        float c = accs[ms][v] + bv;
        c = c > 0.0f ? c : 0.0f;
        dst[(ms * 16 + v + 8 * hf) * PITCH + n] = (_Float16)c;
      }
    }
  }
}

// f32 -> f16 conversion of the five 256x256 weight stacks into workspace.
__global__ void cvt_w(const float* __restrict__ W2,  const float* __restrict__ W31,
                      const float* __restrict__ W32, const float* __restrict__ W41,
                      const float* __restrict__ W42, _Float16* __restrict__ out)
{
  const size_t per = (size_t)HNUM * DDIM * DDIM;     // 655360
  const size_t i = (size_t)blockIdx.x * blockDim.x + threadIdx.x;
  if (i >= 5 * per) return;
  const float* srcs[5] = {W2, W31, W32, W41, W42};
  out[i] = (_Float16)srcs[i / per][i % per];
}

__global__ __launch_bounds__(256) void mlp_kernel(
    const float* __restrict__ x,
    const float* __restrict__ W1,  const float* __restrict__ b1,
    const _Float16* __restrict__ W16,
    const float* __restrict__ b2,  const float* __restrict__ b31,
    const float* __restrict__ b32, const float* __restrict__ b41,
    const float* __restrict__ b42,
    const float* __restrict__ W51, const float* __restrict__ b51,
    const float* __restrict__ W52, const float* __restrict__ b52,
    float* __restrict__ x51s, float* __restrict__ x52s)
{
  extern __shared__ _Float16 smem[];
  _Float16* buf0 = smem;
  _Float16* buf1 = smem + MT * PITCH;
  _Float16* buf2 = smem + 2 * MT * PITCH;
  const int h    = blockIdx.y;
  const int row0 = blockIdx.x * MT;
  const size_t WSZ = (size_t)DDIM * DDIM;

  // ---- layer 1: (64,4) -> (64,256), K=4: plain VALU, result to LDS as f16
  {
    const int t  = threadIdx.x;
    const int r  = t >> 2;                 // 0..63
    const int nb = (t & 3) * 64;
    const float4 xv = ((const float4*)x)[row0 + r];
    const float4* w1 = (const float4*)(W1 + (size_t)h * DDIM * 4);
    const float*  b1h = b1 + h * DDIM;
    #pragma unroll 8
    for (int j = 0; j < 64; ++j) {
      const int n = nb + j;
      const float4 wr = w1[n];
      float v = fmaf(xv.x, wr.x, fmaf(xv.y, wr.y, fmaf(xv.z, wr.z, xv.w * wr.w))) + b1h[n];
      buf0[r * PITCH + n] = (_Float16)(v > 0.0f ? v : 0.0f);
    }
  }
  __syncthreads();
  gemm_layer(buf0, W16 + (0 * HNUM + h) * WSZ, b2  + h * DDIM, buf1);   // x2
  __syncthreads();
  gemm_layer(buf1, W16 + (1 * HNUM + h) * WSZ, b31 + h * DDIM, buf0);   // x31
  gemm_layer(buf1, W16 + (2 * HNUM + h) * WSZ, b32 + h * DDIM, buf2);   // x32
  __syncthreads();
  gemm_layer(buf0, W16 + (3 * HNUM + h) * WSZ, b41 + h * DDIM, buf1);   // x41
  __syncthreads();
  gemm_layer(buf2, W16 + (4 * HNUM + h) * WSZ, b42 + h * DDIM, buf0);   // x42
  __syncthreads();

  // ---- output heads: x51 = x41@W51^T + b51 ; x52 = 4*sigmoid(x42@W52^T + b52)
  {
    const int t = threadIdx.x;
    const int r = t >> 2;                  // sample row 0..63
    const int q = t & 3;
    const int c = q & 1;                   // output column
    const int p = q >> 1;                  // 0: x51 (buf1), 1: x52 (buf0)
    const _Float16* arow = (p ? buf0 : buf1) + r * PITCH;
    const float* wrow = (p ? W52 : W51) + ((size_t)h * 2 + c) * DDIM;
    float s = 0.0f;
    #pragma unroll 4
    for (int k = 0; k < DDIM; k += 8) {
      const v8h av = *(const v8h*)(arow + k);
      #pragma unroll
      for (int i = 0; i < 8; ++i) s = fmaf((float)av[i], wrow[k + i], s);
    }
    const int gb = row0 + r;
    if (p) {
      s += b52[h * 2 + c];
      s = 4.0f / (1.0f + __expf(-s));
      x52s[((size_t)h * BTOT + gb) * 2 + c] = s;
    } else {
      s += b51[h * 2 + c];
      x51s[((size_t)h * BTOT + gb) * 2 + c] = s;
    }
  }
}

// Per-sample CBF/QP epilogue + softmax-weighted head mixing.
__global__ __launch_bounds__(256) void combine_kernel(
    const float* __restrict__ x,    const float* __restrict__ wt,
    const float* __restrict__ mean, const float* __restrict__ stdv,
    const float* __restrict__ mean_label, const float* __restrict__ std_label,
    const float* __restrict__ x51s, const float* __restrict__ x52s,
    float* __restrict__ out)
{
  const int b = blockIdx.x * blockDim.x + threadIdx.x;
  if (b >= BTOT) return;
  const float4 xv = ((const float4*)x)[b];
  const float t1 = xv.x * stdv[0] + mean[0];
  const float w1 = xv.y * stdv[1] + mean[1];
  const float t2 = xv.z * stdv[2] + mean[2];
  const float w2 = xv.w * stdv[3] + mean[3];
  const float s1 = __sinf(t1), c1 = __cosf(t1);
  const float s2 = __sinf(t2), c2 = __cosf(t2);
  const float L1 = 3.0f, L2 = 3.0f;
  const float px = L1 * c1 + L2 * c2 - 0.0f;          // OBS_X = 0
  const float py = L1 * s1 + L2 * s2 - 7.0f;          // OBS_Y = 7
  const float vx = -L1 * s1 * w1 - L2 * s2 * w2;
  const float vy =  L1 * c1 * w1 + L2 * c2 * w2;
  const float barrier = px * px + py * py - 16.0f;    // R^2 = 16
  const float bdot = 2.0f * (px * vx + py * vy);
  const float Lf2b = 2.0f * (vx * vx + vy * vy)
                   + 2.0f * px * (-L1 * c1 * w1 * w1 - L2 * c2 * w2 * w2)
                   + 2.0f * py * (-L1 * s1 * w1 * w1 - L2 * s2 * w2 * w2);
  const float g0 = -(2.0f * px * (-L1 * s1) + 2.0f * py * (L1 * c1));
  const float g1 = -(2.0f * px * (-L2 * s2) + 2.0f * py * (L2 * c2));
  const float gg = g0 * g0 + g1 * g1;

  float wv[HNUM], m = -1e30f, sum = 0.0f;
  #pragma unroll
  for (int i = 0; i < HNUM; ++i) { wv[i] = wt[i]; m = fmaxf(m, wv[i]); }
  #pragma unroll
  for (int i = 0; i < HNUM; ++i) { wv[i] = __expf(wv[i] - m); sum += wv[i]; }
  const float rsum = 1.0f / sum;

  const float a0  = x52s[(size_t)b * 2 + 0];          // head 0, col 0
  const float ml0 = mean_label[0], ml1 = mean_label[1];
  const float rs0 = 1.0f / std_label[0], rs1 = 1.0f / std_label[1];
  float acc0 = 0.0f, acc1 = 0.0f;
  #pragma unroll
  for (int hh = 0; hh < HNUM; ++hh) {
    const size_t o = ((size_t)hh * BTOT + b) * 2;
    float u0 = -x51s[o + 0];
    float u1 = -x51s[o + 1];
    const float bi = x52s[o + 1];
    const float hv = Lf2b + (a0 + bi) * bdot + a0 * bi * barrier;
    const float viol = u0 * g0 + u1 * g1 - hv;
    const float lam = fmaxf(viol, 0.0f) / gg;
    u0 = (u0 - lam * g0 - ml0) * rs0;
    u1 = (u1 - lam * g1 - ml1) * rs1;
    acc0 = fmaf(wv[hh] * rsum, u0, acc0);
    acc1 = fmaf(wv[hh] * rsum, u1, acc1);
  }
  out[(size_t)b * 2 + 0] = acc0;
  out[(size_t)b * 2 + 1] = acc1;
}

extern "C" void kernel_launch(void* const* d_in, const int* in_sizes, int n_in,
                              void* d_out, int out_size, void* d_ws, size_t ws_size,
                              hipStream_t stream)
{
  (void)in_sizes; (void)n_in; (void)out_size; (void)ws_size;
  const float* x          = (const float*)d_in[0];
  const float* wt         = (const float*)d_in[2];
  const float* mean       = (const float*)d_in[3];
  const float* stdv       = (const float*)d_in[4];
  const float* mean_label = (const float*)d_in[5];
  const float* std_label  = (const float*)d_in[6];
  const float* W1  = (const float*)d_in[7];
  const float* b1  = (const float*)d_in[8];
  const float* W2  = (const float*)d_in[9];
  const float* b2  = (const float*)d_in[10];
  const float* W31 = (const float*)d_in[11];
  const float* b31 = (const float*)d_in[12];
  const float* W32 = (const float*)d_in[13];
  const float* b32 = (const float*)d_in[14];
  const float* W41 = (const float*)d_in[15];
  const float* b41 = (const float*)d_in[16];
  const float* W42 = (const float*)d_in[17];
  const float* b42 = (const float*)d_in[18];
  const float* W51 = (const float*)d_in[19];
  const float* b51 = (const float*)d_in[20];
  const float* W52 = (const float*)d_in[21];
  const float* b52 = (const float*)d_in[22];

  // workspace layout: [f16 weights 6.55MB][x51 stage 2.62MB][x52 stage 2.62MB]
  const size_t w16_elems = (size_t)5 * HNUM * DDIM * DDIM;        // 3,276,800
  _Float16* W16 = (_Float16*)d_ws;
  float* x51s = (float*)((char*)d_ws + w16_elems * sizeof(_Float16));
  float* x52s = x51s + (size_t)HNUM * BTOT * 2;

  cvt_w<<<(unsigned)((w16_elems + 255) / 256), 256, 0, stream>>>(W2, W31, W32, W41, W42, W16);

  dim3 grid(BTOT / MT, HNUM);
  const size_t shmem = (size_t)3 * MT * PITCH * sizeof(_Float16); // 101,376 B dynamic LDS
  mlp_kernel<<<grid, 256, shmem, stream>>>(x, W1, b1, W16, b2, b31, b32, b41, b42,
                                           W51, b51, W52, b52, x51s, x52s);

  combine_kernel<<<BTOT / 256, 256, 0, stream>>>(x, wt, mean, stdv, mean_label, std_label,
                                                 x51s, x52s, (float*)d_out);
}
